// RTDETRTransformerV3_8117488190257
// MI455X (gfx1250) — matrix operand order repaired and checked by reading, not compiled, not run on hardware
//
#include <hip/hip_runtime.h>

#define Bn 16
#define Qn 300
#define Dm 256
#define Hn 8
#define HDn 32
#define LVn 3
#define Pp 4
#define NLY 6
#define FFn 1024
#define NCn 80
#define LENm 8400
#define MQ (Bn*Qn)      /* 4800   */
#define MV (Bn*LENm)    /* 134400 */

#define NB 128          /* GEMM col block per workgroup  */
#define MB 128          /* GEMM row block (8 waves x 16) */
#define KC 256          /* GEMM k chunk staged in LDS    */
#define BPITCH (KC + 8) /* LDS row pitch in u16 (bank-conflict pad) */

typedef unsigned short u16;
typedef unsigned int u32;
typedef __attribute__((ext_vector_type(16))) __bf16 v16bf;
typedef __attribute__((ext_vector_type(8)))  float  v8f;

__device__ __forceinline__ u16 f2bf(float f) {
  union { float f; u32 u; } c; c.f = f;
  u32 u = c.u;
  u32 r = u + 0x7FFFu + ((u >> 16) & 1u);   // round-to-nearest-even
  return (u16)(r >> 16);
}
__device__ __forceinline__ float bf2f(u16 h) {
  union { u32 u; float f; } c; c.u = ((u32)h) << 16;
  return c.f;
}

// ---------------- elementwise helpers ----------------
__global__ void k_cvt(const float* __restrict__ x, u16* __restrict__ y, int n) {
  int i = blockIdx.x * 256 + threadIdx.x;
  if (i < n) y[i] = f2bf(x[i]);
}
__global__ void k_f2both(const float* __restrict__ x, float* __restrict__ yf,
                         u16* __restrict__ yb, int n) {
  int i = blockIdx.x * 256 + threadIdx.x;
  if (i < n) { float v = x[i]; yf[i] = v; yb[i] = f2bf(v); }
}
__global__ void k_add(const float* __restrict__ a, const float* __restrict__ b,
                      float* __restrict__ yf, u16* __restrict__ yb, int n) {
  int i = blockIdx.x * 256 + threadIdx.x;
  if (i < n) { float v = a[i] + b[i]; yf[i] = v; yb[i] = f2bf(v); }
}
__global__ void k_sigmoid(const float* __restrict__ x, float* __restrict__ y, int n) {
  int i = blockIdx.x * 256 + threadIdx.x;
  if (i < n) y[i] = 1.f / (1.f + __expf(-x[i]));
}
__global__ void k_refupd(const float* __restrict__ bb, float* __restrict__ ref, int n) {
  int i = blockIdx.x * 256 + threadIdx.x;
  if (i >= n) return;
  float r = fminf(fmaxf(ref[i], 1e-5f), 1.f - 1e-5f);
  float is = __logf(r / (1.f - r));
  ref[i] = 1.f / (1.f + __expf(-(bb[i] + is)));
}

// ---------------- weight pack: W (K,N) fp32 -> Wt (Npad,K) bf16, Npad mult of NB ----
__global__ void k_pack(const float* __restrict__ W, u16* __restrict__ Wt,
                       int K, int N, int Npad) {
  int i = blockIdx.x * 256 + threadIdx.x;
  if (i >= Npad * K) return;
  int n = i / K, k = i - n * K;
  Wt[i] = (n < N) ? f2bf(W[(long)k * N + n]) : (u16)0;
}

// ---------------- WMMA bf16 GEMM: C = A(M,K) @ Wt(Npad,K)^T + bias ----------------
// workgroup = 8 waves = 128x128 output block; B tile staged in LDS per 256-K chunk;
// A fragments loaded direct from global in native WMMA layout. B fragments are
// ping-pong software-pipelined: the ds_load for fragment nt+1 issues before the
// WMMA consuming fragment nt, forcing two live buffers so LDS latency overlaps
// the matrix op instead of serializing behind s_wait_dscnt 0.
union ABfrag { struct { uint4 lo, hi; } q; v16bf v; };

__global__ void __launch_bounds__(256)
k_gemm(const u16* __restrict__ A, const u16* __restrict__ Wt,
       const float* __restrict__ bias, float* __restrict__ Cf,
       u16* __restrict__ Cb, int M, int N, int K, int relu) {
  extern __shared__ u16 Bs[];                 // NB * BPITCH u16
  int t = threadIdx.x;
  int lane = t & 31, wv = t >> 5;
  int r = lane & 15, hh = lane >> 4;
  int m0 = blockIdx.x * MB + wv * 16;
  int n0 = blockIdx.y * NB;
  bool active = (m0 < M);
  const u16* arow = A + (long)(m0 + r) * K;
  v8f acc[8];
#pragma unroll
  for (int i = 0; i < 8; ++i) acc[i] = (v8f){0.f,0.f,0.f,0.f,0.f,0.f,0.f,0.f};

  for (int kc = 0; kc < K; kc += KC) {
    __syncthreads();
    // cooperative stage of B tile: NB rows x KC cols (coalesced b128 -> ds b128)
    for (int idx = t * 8; idx < NB * KC; idx += 256 * 8) {
      int row = idx / KC, col = idx - row * KC;
      uint4 v = *reinterpret_cast<const uint4*>(Wt + (long)(n0 + row) * K + kc + col);
      *reinterpret_cast<uint4*>(&Bs[row * BPITCH + col]) = v;
    }
    __syncthreads();
    if (active) {
      for (int k0 = 0; k0 < KC; k0 += 32) {
        __builtin_prefetch(arow + kc + k0 + 64, 0, 1);
        ABfrag a;
        // A 16x32 fragment: lane half hh -> K runs [k0+8h,+8) and [k0+16+8h,+8)
        a.q.lo = *reinterpret_cast<const uint4*>(arow + kc + k0 + 8 * hh);
        a.q.hi = *reinterpret_cast<const uint4*>(arow + kc + k0 + 16 + 8 * hh);
        const u16* bbase = &Bs[r * BPITCH + k0 + 8 * hh];
        ABfrag bcur, bnext;
        bcur.q.lo = *reinterpret_cast<const uint4*>(bbase);
        bcur.q.hi = *reinterpret_cast<const uint4*>(bbase + 16);
#pragma unroll
        for (int nt = 0; nt < 8; ++nt) {
          if (nt < 7) {
            const u16* bp = bbase + (nt + 1) * 16 * BPITCH;
            bnext.q.lo = *reinterpret_cast<const uint4*>(bp);
            bnext.q.hi = *reinterpret_cast<const uint4*>(bp + 16);
          }
          acc[nt] = __builtin_amdgcn_wmma_f32_16x16x32_bf16(
              false, a.v, false, bcur.v, (short)0, acc[nt], false, false);
          if (nt < 7) bcur = bnext;
        }
      }
    }
  }
  if (!active) return;
#pragma unroll
  for (int nt = 0; nt < 8; ++nt) {
    int col = n0 + nt * 16 + r;
    if (col >= N) continue;
    float bv = bias ? bias[col] : 0.f;
#pragma unroll
    for (int v = 0; v < 8; ++v) {
      int mrow = m0 + v + 8 * hh;            // C layout: VGPR v, lane half hh
      float x = acc[nt][v] + bv;
      if (relu) x = fmaxf(x, 0.f);
      long idx = (long)mrow * N + col;
      if (Cf) Cf[idx] = x;
      if (Cb) Cb[idx] = f2bf(x);
    }
  }
}

// ---------------- layernorm: y = LN(res+dx)*g + b ----------------
__global__ void __launch_bounds__(256)
k_ln(const float* __restrict__ res, const float* __restrict__ dx,
     const float* __restrict__ g, const float* __restrict__ be,
     float* __restrict__ yf, u16* __restrict__ yb) {
  __shared__ float red[8];
  __shared__ float mv[2];
  int row = blockIdx.x, t = threadIdx.x;
  long base = (long)row * Dm;
  float x = res[base + t] + dx[base + t];
  float s = x;
  for (int o = 16; o > 0; o >>= 1) s += __shfl_down(s, o, 32);
  if ((t & 31) == 0) red[t >> 5] = s;
  __syncthreads();
  if (t == 0) { float a = 0; for (int i = 0; i < 8; i++) a += red[i]; mv[0] = a / Dm; }
  __syncthreads();
  float d = x - mv[0];
  float s2 = d * d;
  for (int o = 16; o > 0; o >>= 1) s2 += __shfl_down(s2, o, 32);
  if ((t & 31) == 0) red[t >> 5] = s2;
  __syncthreads();
  if (t == 0) { float a = 0; for (int i = 0; i < 8; i++) a += red[i]; mv[1] = rsqrtf(a / Dm + 1e-5f); }
  __syncthreads();
  float y = d * mv[1] * g[t] + be[t];
  yf[base + t] = y;
  if (yb) yb[base + t] = f2bf(y);
}

// ---------------- MHA core (per b,h), online softmax ----------------
__global__ void __launch_bounds__(256)
k_attn(const float* __restrict__ qp, const float* __restrict__ kp,
       const float* __restrict__ vp, float* __restrict__ out) {
  const int CH = 150;
  __shared__ float Ks[CH * HDn];
  __shared__ float Vs[CH * HDn];
  int bh = blockIdx.x;
  int b = bh >> 3, h = bh & 7;
  long base = (long)b * Qn * Dm + (long)h * HDn;
  const float scale = 0.17677669529663687f;  // 1/sqrt(32)
  float m[2], s[2], acc[2][HDn];
  for (int qi = 0; qi < 2; ++qi) {
    m[qi] = -1e30f; s[qi] = 0.f;
    for (int d = 0; d < HDn; ++d) acc[qi][d] = 0.f;
  }
  for (int c = 0; c < Qn; c += CH) {
    __syncthreads();
    for (int i = threadIdx.x; i < CH * HDn; i += 256) {
      int j = i / HDn, d = i % HDn;
      Ks[i] = kp[base + (long)(c + j) * Dm + d];
      Vs[i] = vp[base + (long)(c + j) * Dm + d];
    }
    __syncthreads();
    for (int qi = 0; qi < 2; ++qi) {
      int q = threadIdx.x + qi * 256;
      if (q >= Qn) continue;
      float qreg[HDn];
      for (int d = 0; d < HDn; ++d) qreg[d] = qp[base + (long)q * Dm + d] * scale;
      for (int j = 0; j < CH; ++j) {
        float dd = 0.f;
        for (int d = 0; d < HDn; ++d) dd += qreg[d] * Ks[j * HDn + d];
        float mn = fmaxf(m[qi], dd);
        float corr = __expf(m[qi] - mn);
        float w = __expf(dd - mn);
        s[qi] = s[qi] * corr + w;
        for (int d = 0; d < HDn; ++d) acc[qi][d] = acc[qi][d] * corr + w * Vs[j * HDn + d];
        m[qi] = mn;
      }
    }
  }
  for (int qi = 0; qi < 2; ++qi) {
    int q = threadIdx.x + qi * 256;
    if (q >= Qn) continue;
    float inv = 1.f / s[qi];
    for (int d = 0; d < HDn; ++d) out[base + (long)q * Dm + d] = acc[qi][d] * inv;
  }
}

// ---------------- qpe hidden: relu(ref @ W1 + b1), K=4, bf16 out ----------------
__global__ void k_qpehid(const float* __restrict__ ref, const float* __restrict__ W1,
                         const float* __restrict__ b1, u16* __restrict__ hid) {
  int i = blockIdx.x * 256 + threadIdx.x;
  if (i >= MQ * 2 * Dm) return;
  int row = i / (2 * Dm), j = i - row * (2 * Dm);
  const float* r = ref + row * 4;
  float a = b1[j] + r[0] * W1[j] + r[1] * W1[2 * Dm + j]
          + r[2] * W1[2 * 2 * Dm + j] + r[3] * W1[3 * 2 * Dm + j];
  hid[i] = f2bf(fmaxf(a, 0.f));
}

// ---------------- MSDA sampling: grid = (b*Q), thread = (h,d) ----------------
__global__ void __launch_bounds__(256)
k_msda(const float* __restrict__ off, const float* __restrict__ awl,
       const float* __restrict__ ref, const u16* __restrict__ val,
       u16* __restrict__ outB) {
  int row = blockIdx.x;            // b*Q + q
  int b = row / Qn;
  int t = threadIdx.x;
  int h = t >> 5, d = t & 31;
  const int Hls[3] = {80, 40, 20}, Wls[3] = {80, 40, 20}, st[3] = {0, 6400, 8000};
  const float* o  = off + (long)row * (Hn * LVn * Pp * 2) + h * (LVn * Pp * 2);
  const float* aw = awl + (long)row * (Hn * LVn * Pp)     + h * (LVn * Pp);
  float mx = -1e30f;
  for (int i = 0; i < 12; i++) mx = fmaxf(mx, aw[i]);
  float w[12], se = 0.f;
  for (int i = 0; i < 12; i++) { w[i] = __expf(aw[i] - mx); se += w[i]; }
  float inv = 1.f / se;
  float rx = ref[row * 4 + 0], ry = ref[row * 4 + 1];
  float rw = ref[row * 4 + 2], rh = ref[row * 4 + 3];
  float acc = 0.f;
  for (int l = 0; l < 3; ++l) {
    int Hl = Hls[l], Wl = Wls[l];
    long vbase = ((long)b * LENm + st[l]) * Dm + h * HDn + d;
    for (int p = 0; p < 4; ++p) {
      float ox = o[(l * 4 + p) * 2 + 0], oy = o[(l * 4 + p) * 2 + 1];
      float lx = rx + ox * 0.25f * rw * 0.5f;
      float ly = ry + oy * 0.25f * rh * 0.5f;
      float x = lx * Wl - 0.5f, y = ly * Hl - 0.5f;
      float x0f = floorf(x), y0f = floorf(y);
      int x0 = (int)x0f, y0 = (int)y0f;
      float wx1 = x - x0f, wy1 = y - y0f;
      float wx0 = 1.f - wx1, wy0 = 1.f - wy1;
      bool xi0 = (x0 >= 0) && (x0 < Wl), xi1 = (x0 + 1 >= 0) && (x0 + 1 < Wl);
      bool yi0 = (y0 >= 0) && (y0 < Hl), yi1 = (y0 + 1 >= 0) && (y0 + 1 < Hl);
      float v00 = 0.f, v01 = 0.f, v10 = 0.f, v11 = 0.f;
      if (yi0) {
        long rb = vbase + (long)y0 * Wl * Dm;
        if (xi0) v00 = bf2f(val[rb + (long)x0 * Dm]);
        if (xi1) v01 = bf2f(val[rb + (long)(x0 + 1) * Dm]);
      }
      if (yi1) {
        long rb = vbase + (long)(y0 + 1) * Wl * Dm;
        if (xi0) v10 = bf2f(val[rb + (long)x0 * Dm]);
        if (xi1) v11 = bf2f(val[rb + (long)(x0 + 1) * Dm]);
      }
      float sv = (v00 * wx0 + v01 * wx1) * wy0 + (v10 * wx0 + v11 * wx1) * wy1;
      acc += sv * (w[l * 4 + p] * inv);
    }
  }
  outB[(long)row * Dm + t] = f2bf(acc);
}

// =====================================================================
extern "C" void kernel_launch(void* const* d_in, const int* in_sizes, int n_in,
                              void* d_out, int out_size, void* d_ws, size_t ws_size,
                              hipStream_t stream) {
  (void)in_sizes; (void)n_in; (void)out_size; (void)ws_size;
  const float* tgt  = (const float*)d_in[0];
  const float* refu = (const float*)d_in[1];
  const float* mem  = (const float*)d_in[2];
  const float* sWq = (const float*)d_in[4];  const float* sbq = (const float*)d_in[5];
  const float* sWk = (const float*)d_in[6];  const float* sbk = (const float*)d_in[7];
  const float* sWv = (const float*)d_in[8];  const float* sbv = (const float*)d_in[9];
  const float* sWo = (const float*)d_in[10]; const float* sbo = (const float*)d_in[11];
  const float* ln1g = (const float*)d_in[12]; const float* ln1b = (const float*)d_in[13];
  const float* ln2g = (const float*)d_in[14]; const float* ln2b = (const float*)d_in[15];
  const float* ln3g = (const float*)d_in[16]; const float* ln3b = (const float*)d_in[17];
  const float* offW = (const float*)d_in[18]; const float* offb = (const float*)d_in[19];
  const float* awW  = (const float*)d_in[20]; const float* awb  = (const float*)d_in[21];
  const float* valW = (const float*)d_in[22]; const float* valb = (const float*)d_in[23];
  const float* outW = (const float*)d_in[24]; const float* outb = (const float*)d_in[25];
  const float* f1W  = (const float*)d_in[26]; const float* f1b  = (const float*)d_in[27];
  const float* f2W  = (const float*)d_in[28]; const float* f2b  = (const float*)d_in[29];
  const float* q1W  = (const float*)d_in[30]; const float* q1b  = (const float*)d_in[31];
  const float* q2W  = (const float*)d_in[32]; const float* q2b  = (const float*)d_in[33];
  const float* b1W  = (const float*)d_in[34]; const float* b1b  = (const float*)d_in[35];
  const float* b2W  = (const float*)d_in[36]; const float* b2b  = (const float*)d_in[37];
  const float* b3W  = (const float*)d_in[38]; const float* b3b  = (const float*)d_in[39];
  const float* scW  = (const float*)d_in[40]; const float* scb  = (const float*)d_in[41];

  char* base = (char*)d_ws; size_t cur = 0;
  auto alloc = [&](size_t bytes) -> void* {
    cur = (cur + 255) & ~(size_t)255;
    void* p = base + cur; cur += bytes; return p;
  };
  u16*   memB  = (u16*)  alloc((size_t)MV * Dm * 2);
  u16*   valB  = (u16*)  alloc((size_t)MV * Dm * 2);
  float* outF  = (float*)alloc((size_t)MQ * Dm * 4);
  u16*   outBq = (u16*)  alloc((size_t)MQ * Dm * 2);
  float* qpeF  = (float*)alloc((size_t)MQ * Dm * 4);
  float* qF    = (float*)alloc((size_t)MQ * Dm * 4);
  u16*   qB    = (u16*)  alloc((size_t)MQ * Dm * 2);
  u16*   hidQ  = (u16*)  alloc((size_t)MQ * 2 * Dm * 2);
  float* qpF   = (float*)alloc((size_t)MQ * Dm * 4);
  float* kpF   = (float*)alloc((size_t)MQ * Dm * 4);
  float* vpF   = (float*)alloc((size_t)MQ * Dm * 4);
  float* attF  = (float*)alloc((size_t)MQ * Dm * 4);
  u16*   attB  = (u16*)  alloc((size_t)MQ * Dm * 2);
  float* mhaF  = (float*)alloc((size_t)MQ * Dm * 4);
  float* offF  = (float*)alloc((size_t)MQ * 192 * 4);
  float* awF   = (float*)alloc((size_t)MQ * 96 * 4);
  u16*   sampB = (u16*)  alloc((size_t)MQ * Dm * 2);
  float* msdaF = (float*)alloc((size_t)MQ * Dm * 4);
  u16*   ffnhB = (u16*)  alloc((size_t)MQ * FFn * 2);
  float* ffn2F = (float*)alloc((size_t)MQ * Dm * 4);
  u16*   bb1B  = (u16*)  alloc((size_t)MQ * Dm * 2);
  u16*   bb2B  = (u16*)  alloc((size_t)MQ * Dm * 2);
  float* bb3F  = (float*)alloc((size_t)MQ * 4 * 4);
  float* refF  = (float*)alloc((size_t)MQ * 4 * 4);
  // packed weights: Npad = round-up to 128 rows
  u16* wqT    = (u16*)alloc((size_t)256  * Dm * 2);
  u16* wkT    = (u16*)alloc((size_t)256  * Dm * 2);
  u16* wvT    = (u16*)alloc((size_t)256  * Dm * 2);
  u16* woT    = (u16*)alloc((size_t)256  * Dm * 2);
  u16* wvalT  = (u16*)alloc((size_t)256  * Dm * 2);
  u16* woutT  = (u16*)alloc((size_t)256  * Dm * 2);
  u16* wbb1T  = (u16*)alloc((size_t)256  * Dm * 2);
  u16* wbb2T  = (u16*)alloc((size_t)256  * Dm * 2);
  u16* woffT  = (u16*)alloc((size_t)256  * Dm * 2);   // N=192 -> 256
  u16* wawT   = (u16*)alloc((size_t)128  * Dm * 2);   // N=96  -> 128
  u16* wffn1T = (u16*)alloc((size_t)1024 * Dm * 2);
  u16* wffn2T = (u16*)alloc((size_t)256  * FFn * 2);
  u16* wbb3T  = (u16*)alloc((size_t)128  * Dm * 2);   // N=4   -> 128
  u16* wscT   = (u16*)alloc((size_t)128  * Dm * 2);   // N=80  -> 128
  u16* wq2T   = (u16*)alloc((size_t)256  * 2 * Dm * 2);

  dim3 blk256(256);
  auto cdiv = [](long a, long b) { return (unsigned)((a + b - 1) / b); };
  auto pack = [&](const float* W, u16* Wt, int K, int N) {
    int Np = (N + NB - 1) & ~(NB - 1);
    k_pack<<<dim3(cdiv((long)Np * K, 256)), blk256, 0, stream>>>(W, Wt, K, N, Np);
  };
  const size_t shm = (size_t)NB * BPITCH * sizeof(u16);   // ~66 KB
  auto gemm = [&](const u16* A, const u16* Wt, const float* bias,
                  float* Cf, u16* Cb, int M, int N, int K, int relu) {
    int Np = (N + NB - 1) & ~(NB - 1);
    dim3 g((M + MB - 1) / MB, Np / NB);
    k_gemm<<<g, blk256, shm, stream>>>(A, Wt, bias, Cf, Cb, M, N, K, relu);
  };

  // one-time prep
  k_cvt<<<dim3(cdiv((long)MV * Dm, 256)), blk256, 0, stream>>>(mem, memB, MV * Dm);
  pack(q2W, wq2T, 2 * Dm, Dm);
  k_sigmoid<<<dim3(cdiv(MQ * 4, 256)), blk256, 0, stream>>>(refu, refF, MQ * 4);
  k_f2both<<<dim3(cdiv((long)MQ * Dm, 256)), blk256, 0, stream>>>(tgt, outF, outBq, MQ * Dm);

  for (int i = 0; i < NLY; ++i) {
    // query positional embedding
    k_qpehid<<<dim3(cdiv((long)MQ * 2 * Dm, 256)), blk256, 0, stream>>>(refF, q1W, q1b, hidQ);
    gemm(hidQ, wq2T, q2b, qpeF, nullptr, MQ, Dm, 2 * Dm, 0);
    // self-attention
    k_add<<<dim3(cdiv((long)MQ * Dm, 256)), blk256, 0, stream>>>(outF, qpeF, qF, qB, MQ * Dm);
    pack(sWq + (size_t)i * Dm * Dm, wqT, Dm, Dm);
    pack(sWk + (size_t)i * Dm * Dm, wkT, Dm, Dm);
    pack(sWv + (size_t)i * Dm * Dm, wvT, Dm, Dm);
    pack(sWo + (size_t)i * Dm * Dm, woT, Dm, Dm);
    gemm(qB,    wqT, sbq + i * Dm, qpF, nullptr, MQ, Dm, Dm, 0);
    gemm(qB,    wkT, sbk + i * Dm, kpF, nullptr, MQ, Dm, Dm, 0);
    gemm(outBq, wvT, sbv + i * Dm, vpF, nullptr, MQ, Dm, Dm, 0);
    k_attn<<<dim3(Bn * Hn), blk256, 0, stream>>>(qpF, kpF, vpF, attF);
    k_cvt<<<dim3(cdiv((long)MQ * Dm, 256)), blk256, 0, stream>>>(attF, attB, MQ * Dm);
    gemm(attB, woT, sbo + i * Dm, mhaF, nullptr, MQ, Dm, Dm, 0);
    k_ln<<<dim3(MQ), blk256, 0, stream>>>(outF, mhaF, ln1g + i * Dm, ln1b + i * Dm, outF, outBq);
    // value projection (dominant GEMM; bf16 memory tile stays resident in L2)
    pack(valW + (size_t)i * Dm * Dm, wvalT, Dm, Dm);
    gemm(memB, wvalT, valb + i * Dm, nullptr, valB, MV, Dm, Dm, 0);
    // MSDA
    k_add<<<dim3(cdiv((long)MQ * Dm, 256)), blk256, 0, stream>>>(outF, qpeF, qF, qB, MQ * Dm);
    pack(offW + (size_t)i * Dm * 192, woffT, Dm, 192);
    pack(awW  + (size_t)i * Dm * 96,  wawT,  Dm, 96);
    gemm(qB, woffT, offb + i * 192, offF, nullptr, MQ, 192, Dm, 0);
    gemm(qB, wawT,  awb  + i * 96,  awF,  nullptr, MQ, 96,  Dm, 0);
    k_msda<<<dim3(MQ), blk256, 0, stream>>>(offF, awF, refF, valB, sampB);
    pack(outW + (size_t)i * Dm * Dm, woutT, Dm, Dm);
    gemm(sampB, woutT, outb + i * Dm, msdaF, nullptr, MQ, Dm, Dm, 0);
    k_ln<<<dim3(MQ), blk256, 0, stream>>>(outF, msdaF, ln2g + i * Dm, ln2b + i * Dm, outF, outBq);
    // FFN
    pack(f1W + (size_t)i * Dm * FFn, wffn1T, Dm, FFn);
    pack(f2W + (size_t)i * FFn * Dm, wffn2T, FFn, Dm);
    gemm(outBq, wffn1T, f1b + i * FFn, nullptr, ffnhB, MQ, FFn, Dm, 1);
    gemm(ffnhB, wffn2T, f2b + i * Dm,  ffn2F, nullptr, MQ, Dm, FFn, 0);
    k_ln<<<dim3(MQ), blk256, 0, stream>>>(outF, ffn2F, ln3g + i * Dm, ln3b + i * Dm, outF, outBq);
    // bbox head + ref update
    pack(b1W + (size_t)i * Dm * Dm, wbb1T, Dm, Dm);
    pack(b2W + (size_t)i * Dm * Dm, wbb2T, Dm, Dm);
    pack(b3W + (size_t)i * Dm * 4,  wbb3T, Dm, 4);
    gemm(outBq, wbb1T, b1b + i * Dm, nullptr, bb1B, MQ, Dm, Dm, 1);
    gemm(bb1B,  wbb2T, b2b + i * Dm, nullptr, bb2B, MQ, Dm, Dm, 1);
    gemm(bb2B,  wbb3T, b3b + i * 4,  bb3F, nullptr, MQ, 4, Dm, 0);
    k_refupd<<<dim3(cdiv(MQ * 4, 256)), blk256, 0, stream>>>(bb3F, refF, MQ * 4);
  }

  // final logits from last layer, written straight into d_out after the ref slot
  pack(scW + (size_t)(NLY - 1) * Dm * NCn, wscT, Dm, NCn);
  gemm(outBq, wscT, scb + (NLY - 1) * NCn, (float*)d_out + (size_t)MQ * 4, nullptr,
       MQ, NCn, Dm, 0);
  hipMemcpyAsync(d_out, refF, (size_t)MQ * 4 * sizeof(float),
                 hipMemcpyDeviceToDevice, stream);
}
